// CausalSelfAttention_292057776332
// MI455X (gfx1250) — matrix operand order repaired and checked by reading, not compiled, not run on hardware
//
#include <hip/hip_runtime.h>

// ---------------------------------------------------------------------------
// CausalSelfAttention for MI455X (gfx1250), bf16 WMMA pipeline, fp32 accum.
// B=2, T=2048, C=1024, H=16, D=64.
//   - weights pre-transposed so GEMM A and B tiles stage with contiguous B128
//   - async global->LDS copies (ASYNCcnt) + double-buffered K slabs (BK=64)
//   - flash attention with online softmax, causal-bounded key loop
// ---------------------------------------------------------------------------

typedef unsigned short u16;
typedef unsigned int   u32;
typedef __attribute__((ext_vector_type(16))) __bf16 v16bf;
typedef __attribute__((ext_vector_type(8)))  float  v8f;
typedef __attribute__((ext_vector_type(4)))  int    v4i;

#if __has_builtin(__builtin_amdgcn_global_load_async_to_lds_b128)
#define HAVE_ASYNC_LDS 1
#else
#define HAVE_ASYNC_LDS 0
#endif

union Frag16 {            // one WMMA 16-bit A/B fragment: 16 bf16 = 32 bytes
    v16bf v;
    uint4 q[2];
    u16   s[16];
};

__device__ __forceinline__ u16 f2bf(float f) {
    union { float f; u32 u; } x; x.f = f;
    u32 r = x.u + 0x7FFFu + ((x.u >> 16) & 1u);   // round-to-nearest-even
    return (u16)(r >> 16);
}

__device__ __forceinline__ v8f wmma_bf16(const v16bf& a, const v16bf& b, const v8f& c) {
    return __builtin_amdgcn_wmma_f32_16x16x32_bf16(false, a, false, b, (short)0, c,
                                                   false, false);
}

__device__ __forceinline__ void copy16_lds(u16* lp, const u16* gp) {
#if HAVE_ASYNC_LDS
    __builtin_amdgcn_global_load_async_to_lds_b128(
        (__attribute__((address_space(1))) v4i*)(v4i*)gp,
        (__attribute__((address_space(3))) v4i*)(v4i*)lp, 0, 0);
#else
    *(uint4*)lp = *(const uint4*)gp;
#endif
}

__device__ __forceinline__ void wait_async_lds() {
#if HAVE_ASYNC_LDS
#if __has_builtin(__builtin_amdgcn_s_wait_asynccnt)
    __builtin_amdgcn_s_wait_asynccnt(0);
#else
    asm volatile("s_wait_asynccnt 0x0" ::: "memory");
#endif
#endif
}

// ---------------------------------------------------------------------------
// f32 -> bf16 conversion (grid-stride)
// ---------------------------------------------------------------------------
__global__ void cvt_f32_bf16_kernel(const float* __restrict__ src,
                                    u16* __restrict__ dst, int n) {
    for (int i = blockIdx.x * blockDim.x + threadIdx.x; i < n;
         i += gridDim.x * blockDim.x)
        dst[i] = f2bf(src[i]);
}

// ---------------------------------------------------------------------------
// f32 [K][N] -> bf16 transposed [N][K] (tiled, coalesced both sides)
// ---------------------------------------------------------------------------
__global__ __launch_bounds__(256)
void cvt_transpose_bf16_kernel(const float* __restrict__ W, u16* __restrict__ Wt,
                               int K, int N) {
    __shared__ float tile[32][33];
    const int bx = blockIdx.x;           // along N
    const int by = blockIdx.y;           // along K
    const int tx = threadIdx.x & 31;
    const int ty = threadIdx.x >> 5;     // 0..7
#pragma unroll
    for (int r = 0; r < 32; r += 8)
        tile[r + ty][tx] = W[(size_t)(by * 32 + r + ty) * N + bx * 32 + tx];
    __syncthreads();
#pragma unroll
    for (int r = 0; r < 32; r += 8)
        Wt[(size_t)(bx * 32 + r + ty) * K + by * 32 + tx] = f2bf(tile[tx][r + ty]);
}

// ---------------------------------------------------------------------------
// bf16 WMMA GEMM: C[M,N] = A[M,K] * Bt[N,K]^T + bias[N]
//   block = 128x128, 256 threads (8 waves), wave = 32x64 (2x4 WMMA tiles)
//   BK = 64, double-buffered LDS slabs staged with async global->LDS B128
//   mode 0: scatter qkv (N=3072) bf16 -> q[BH,T,D], k[BH,T,D], vT[BH,D,T]
//   mode 1: fp32 out[M,N]
// ---------------------------------------------------------------------------
__global__ __launch_bounds__(256)
void gemm_bf16_wmma_kernel(const u16* __restrict__ A, const u16* __restrict__ Bt,
                           const float* __restrict__ bias,
                           float* __restrict__ outF,
                           u16* __restrict__ outQ, u16* __restrict__ outK,
                           u16* __restrict__ outVt,
                           int M, int N, int K, int mode) {
    // [buf][A/B][128 rows][72] (64 + 8 pad, rows 144B -> 16B aligned)
    __shared__ u16 lds[2][2][128 * 72];

    const int tid  = threadIdx.x;
    const int lane = tid & 31;
    const int w    = tid >> 5;
    const int wm   = w & 3;          // wave m-offset: 32*wm
    const int wn   = w >> 2;         // wave n-offset: 64*wn
    const int m0   = blockIdx.y * 128;
    const int n0   = blockIdx.x * 128;

    const int halfk8  = (lane >> 4) * 8;    // A-frag k base per lane-half
    const int halfk16 = (lane >> 4) * 16;   // unused; B frags also use halfk8
    const int l15     = lane & 15;

    v8f acc[2][4];
#pragma unroll
    for (int mt = 0; mt < 2; ++mt)
#pragma unroll
        for (int nt = 0; nt < 4; ++nt) acc[mt][nt] = {};

    const u16* aBase = A  + (size_t)m0 * K;   // 128 rows x K
    const u16* bBase = Bt + (size_t)n0 * K;   // 128 rows x K

    // cooperative 128x64 tile stage: 1024 chunks of 8 elems, 4 per thread
    auto stage = [&](int buf, const u16* src, int ab, int k0) {
        u16* dst = &lds[buf][ab][0];
#pragma unroll
        for (int p = 0; p < 4; ++p) {
            int cl = p * 256 + tid;
            int r = cl >> 3, c = (cl & 7) * 8;
            copy16_lds(dst + r * 72 + c, src + (size_t)r * K + k0 + c);
        }
    };

    const int NIT = K >> 6;
    stage(0, aBase, 0, 0);
    stage(0, bBase, 1, 0);

    for (int it = 0; it < NIT; ++it) {
        wait_async_lds();          // own wave's slab copies done
        __syncthreads();           // all waves' copies visible; prev compute done
        if (it + 1 < NIT) {        // prefetch next slab into the other buffer
            int kn = (it + 1) << 6;
            stage((it + 1) & 1, aBase, 0, kn);
            stage((it + 1) & 1, bBase, 1, kn);
        }
        const u16* As = &lds[it & 1][0][0];
        const u16* Bs = &lds[it & 1][1][0];
#pragma unroll
        for (int kc = 0; kc < 2; ++kc) {       // two K=32 chunks per slab
            Frag16 af[2];
#pragma unroll
            for (int mt = 0; mt < 2; ++mt) {
                const u16* base = As + (wm * 32 + mt * 16 + l15) * 72 + kc * 32;
                af[mt].q[0] = *(const uint4*)(base + halfk8);
                af[mt].q[1] = *(const uint4*)(base + halfk8 + 16);
            }
#pragma unroll
            for (int nt = 0; nt < 4; ++nt) {
                const u16* base = Bs + (wn * 64 + nt * 16 + l15) * 72 + kc * 32 + halfk8;
                Frag16 bf;
                bf.q[0] = *(const uint4*)(base);
                bf.q[1] = *(const uint4*)(base + 16);
#pragma unroll
                for (int mt = 0; mt < 2; ++mt)
                    acc[mt][nt] = wmma_bf16(af[mt].v, bf.v, acc[mt][nt]);
            }
        }
    }

    // ---- epilogue: C/D layout -> M = e + 8*(lane>=16), N = lane&15 ----
    const int mhalf = (lane >> 4) * 8;
#pragma unroll
    for (int mt = 0; mt < 2; ++mt) {
#pragma unroll
        for (int nt = 0; nt < 4; ++nt) {
#pragma unroll
            for (int e = 0; e < 8; ++e) {
                int m = m0 + wm * 32 + mt * 16 + e + mhalf;
                int n = n0 + wn * 64 + nt * 16 + l15;
                float val = acc[mt][nt][e] + bias[n];
                if (mode == 1) {
                    outF[(size_t)m * N + n] = val;
                } else {
                    int sel = n >> 10, nn = n & 1023;
                    int h = nn >> 6, d = nn & 63;
                    int b = m >> 11, t = m & 2047;
                    int bh = b * 16 + h;
                    u16 bv = f2bf(val);
                    if (sel == 0)      outQ[((size_t)bh * 2048 + t) * 64 + d] = bv;
                    else if (sel == 1) outK[((size_t)bh * 2048 + t) * 64 + d] = bv;
                    else               outVt[((size_t)bh * 64 + d) * 2048 + t] = bv;
                }
            }
        }
    }
}

// ---------------------------------------------------------------------------
// Flash attention: 8 waves/block, each wave owns 16 query rows.
//   Q,K bf16 [BH][T][64], V bf16 transposed [BH][64][T], Y bf16 [B*T][1024].
// ---------------------------------------------------------------------------
__global__ __launch_bounds__(256)
void attn_flash_wmma_kernel(const u16* __restrict__ Q, const u16* __restrict__ Kv,
                            const u16* __restrict__ Vt, u16* __restrict__ Y) {
    __shared__ u16 plds[8][16 * 72];   // per-wave P patch, padded stride 72

    const int lane  = threadIdx.x & 31;
    const int w     = threadIdx.x >> 5;
    const int qblk  = blockIdx.x & 15;      // 16 q-blocks of 128 per (b,h)
    const int bh    = blockIdx.x >> 4;      // 0..31
    const int b     = bh >> 4, h = bh & 15;
    const int qbase = qblk * 128 + w * 16;

    const u16* qp = Q  + (size_t)bh * 2048 * 64;
    const u16* kp = Kv + (size_t)bh * 2048 * 64;
    const u16* vp = Vt + (size_t)bh * 64 * 2048;

    const int l15 = lane & 15;
    const int h8  = (lane >> 4) * 8;        // row half / A-frag k base
    const int h16 = (lane >> 4) * 16;       // unused; B frags also use h8

    Frag16 qf[2];
    {
        const u16* base = qp + (size_t)(qbase + l15) * 64;
#pragma unroll
        for (int s = 0; s < 2; ++s) {
            qf[s].q[0] = *(const uint4*)(base + s * 32 + h8);
            qf[s].q[1] = *(const uint4*)(base + s * 32 + h8 + 16);
        }
    }

    v8f O[4];
#pragma unroll
    for (int i = 0; i < 4; ++i) O[i] = {};
    float mi[8], li[8];
#pragma unroll
    for (int e = 0; e < 8; ++e) { mi[e] = -1e30f; li[e] = 0.0f; }

    const float scale = 0.125f;             // 1/sqrt(64)
    const int qend = qbase + 15;
    u16* pw = &plds[w][0];

    for (int j0 = 0; j0 <= qend; j0 += 64) {
        // ---- S = Q * K^T for 64 keys (4 n-tiles x 2 k-chunks) ----
        v8f S[4];
#pragma unroll
        for (int nt = 0; nt < 4; ++nt) {
            S[nt] = {};
            const u16* kbase = kp + (size_t)(j0 + nt * 16 + l15) * 64 + h8;
#pragma unroll
            for (int s = 0; s < 2; ++s) {
                Frag16 kf;
                kf.q[0] = *(const uint4*)(kbase + s * 32);
                kf.q[1] = *(const uint4*)(kbase + s * 32 + 16);
                S[nt] = wmma_bf16(qf[s].v, kf.v, S[nt]);
            }
        }
        // ---- online softmax ----
#pragma unroll
        for (int e = 0; e < 8; ++e) {
            int qi = qbase + e + h8;
#pragma unroll
            for (int nt = 0; nt < 4; ++nt) {
                int j = j0 + nt * 16 + l15;
                float sv = S[nt][e] * scale;
                if (j > qi) sv = -1e30f;     // causal mask
                S[nt][e] = sv;
            }
            float rmax = fmaxf(fmaxf(S[0][e], S[1][e]), fmaxf(S[2][e], S[3][e]));
            rmax = fmaxf(rmax, __shfl_xor(rmax, 1));
            rmax = fmaxf(rmax, __shfl_xor(rmax, 2));
            rmax = fmaxf(rmax, __shfl_xor(rmax, 4));
            rmax = fmaxf(rmax, __shfl_xor(rmax, 8));
            float mnew  = fmaxf(mi[e], rmax);
            float alpha = __expf(mi[e] - mnew);
            float rsum = 0.0f;
#pragma unroll
            for (int nt = 0; nt < 4; ++nt) {
                float p = __expf(S[nt][e] - mnew);
                S[nt][e] = p;
                rsum += p;
            }
            rsum += __shfl_xor(rsum, 1);
            rsum += __shfl_xor(rsum, 2);
            rsum += __shfl_xor(rsum, 4);
            rsum += __shfl_xor(rsum, 8);
            li[e] = li[e] * alpha + rsum;
            mi[e] = mnew;
#pragma unroll
            for (int nd = 0; nd < 4; ++nd) O[nd][e] *= alpha;
        }
        // ---- C/D-layout P -> A-fragment layout via per-wave LDS patch ----
#pragma unroll
        for (int nt = 0; nt < 4; ++nt)
#pragma unroll
            for (int e = 0; e < 8; ++e)
                pw[(e + h8) * 72 + nt * 16 + l15] = f2bf(S[nt][e]);
        Frag16 pf[2];
        {
            const u16* base = pw + l15 * 72;
#pragma unroll
            for (int s = 0; s < 2; ++s) {
                pf[s].q[0] = *(const uint4*)(base + s * 32 + h8);
                pf[s].q[1] = *(const uint4*)(base + s * 32 + h8 + 16);
            }
        }
        // ---- O += P * V  (V^T layout -> contiguous B-frags) ----
#pragma unroll
        for (int nd = 0; nd < 4; ++nd) {
            const u16* vbase = vp + (size_t)(nd * 16 + l15) * 2048 + j0 + h8;
#pragma unroll
            for (int s = 0; s < 2; ++s) {
                Frag16 vf;
                vf.q[0] = *(const uint4*)(vbase + s * 32);
                vf.q[1] = *(const uint4*)(vbase + s * 32 + 16);
                O[nd] = wmma_bf16(pf[s].v, vf.v, O[nd]);
            }
        }
    }

    // ---- finalize: y[b*T + t][h*64 + d] = O / l ----
#pragma unroll
    for (int nd = 0; nd < 4; ++nd) {
#pragma unroll
        for (int e = 0; e < 8; ++e) {
            int t   = qbase + e + h8;
            int col = h * 64 + nd * 16 + l15;
            float v = O[nd][e] / li[e];
            Y[((size_t)(b * 2048 + t)) * 1024 + col] = f2bf(v);
        }
    }
}

// ---------------------------------------------------------------------------
// Host launcher
// ---------------------------------------------------------------------------
extern "C" void kernel_launch(void* const* d_in, const int* in_sizes, int n_in,
                              void* d_out, int out_size, void* d_ws, size_t ws_size,
                              hipStream_t stream) {
    (void)in_sizes; (void)n_in; (void)out_size; (void)ws_size;
    const float* x  = (const float*)d_in[0];   // [2,2048,1024]
    const float* Wa = (const float*)d_in[1];   // [1024,3072]
    const float* ba = (const float*)d_in[2];   // [3072]
    const float* Wp = (const float*)d_in[3];   // [1024,1024]
    const float* bp = (const float*)d_in[4];   // [1024]
    float* out = (float*)d_out;                // [2,2048,1024] fp32

    const size_t MB = 1u << 20;
    char* ws = (char*)d_ws;
    u16* xb  = (u16*)(ws);             // x    bf16 [4096][1024]   8 MB
    u16* wab = (u16*)(ws + 8 * MB);    // Wa^T bf16 [3072][1024]   6 MB
    u16* wpb = (u16*)(ws + 14 * MB);   // Wp^T bf16 [1024][1024]   2 MB
    u16* qb  = (u16*)(ws + 16 * MB);   // q    bf16 [32][2048][64] 8 MB
    u16* kb  = (u16*)(ws + 24 * MB);   // k    bf16 [32][2048][64] 8 MB
    u16* vbt = (u16*)(ws + 32 * MB);   // v^T  bf16 [32][64][2048] 8 MB
    u16* yb  = (u16*)(ws + 40 * MB);   // y    bf16 [4096][1024]   8 MB

    cvt_f32_bf16_kernel<<<2048, 256, 0, stream>>>(x, xb, 4096 * 1024);
    cvt_transpose_bf16_kernel<<<dim3(3072 / 32, 1024 / 32), 256, 0, stream>>>(
        Wa, wab, 1024, 3072);
    cvt_transpose_bf16_kernel<<<dim3(1024 / 32, 1024 / 32), 256, 0, stream>>>(
        Wp, wpb, 1024, 1024);

    // qkv = x @ W_attn + b_attn  -> scatter q/k/v^T (bf16)
    gemm_bf16_wmma_kernel<<<dim3(3072 / 128, 4096 / 128), 256, 0, stream>>>(
        xb, wab, ba, nullptr, qb, kb, vbt, 4096, 3072, 1024, 0);

    // flash attention -> y bf16 [B*T][C]
    attn_flash_wmma_kernel<<<32 * 16, 256, 0, stream>>>(qb, kb, vbt, yb);

    // out = y @ W_proj + b_proj (fp32)
    gemm_bf16_wmma_kernel<<<dim3(1024 / 128, 4096 / 128), 256, 0, stream>>>(
        yb, wpb, bp, out, nullptr, nullptr, nullptr, 4096, 1024, 1024, 1);
}
